// MultiHeadAttention_37211596653299
// MI455X (gfx1250) — matrix operand order repaired and checked by reading, not compiled, not run on hardware
//
#include <hip/hip_runtime.h>

// MI455X (gfx1250) MHA forward. wave32, WMMA bf16 16x16x32 everywhere.
// B=2, S=2048, D=1024, H=16, HD=64.
// Pipeline: cvt weights -> QKV proj (WMMA; V written transposed) ->
// fully fused scores+softmax+PV (WMMA; the 537MB attn_weights tensor is
// written exactly once and never re-read from HBM) -> output proj (WMMA).

typedef __attribute__((ext_vector_type(16))) __bf16 v16bf;
typedef __attribute__((ext_vector_type(8)))  float  v8f;

#define S_LEN 2048
#define D_DIM 1024
#define NHEAD 16
#define HDIM  64
#define BH    32              // B*H
#define MROWS 4096            // B*S

#define BM 128
#define BN 64
#define BK 32
#define LDA 34                // BK + 2 pad (keeps 4B alignment)
#define LDB 34
#define LDP 1032              // 1024 + 8 pad for staged P rows

__device__ __forceinline__ unsigned short f2bf(float f) {
  union { float f; unsigned int u; } v; v.f = f;
  unsigned int r = v.u + 0x7FFFu + ((v.u >> 16) & 1u);   // round-to-nearest-even
  return (unsigned short)(r >> 16);
}

union FragBF { v16bf v; unsigned int u[8]; };

// A fragment (16x32 bf16, M x K). src points to row 0 of the 16-row slice,
// row-major with leading dim ld (ushorts). lane&15 = M, lane>>4 = K-half.
// ISA layout: VGPR j<4 -> K = half*8 + 2j ; j>=4 -> K = 16 + half*8 + 2(j-4).
__device__ __forceinline__ v16bf frag_a_from(const unsigned short* src, int ld, int lane) {
  const int m = lane & 15, hlf = lane >> 4;
  const unsigned short* row = src + m * ld;
  FragBF f;
#pragma unroll
  for (int j = 0; j < 8; ++j) {
    int k = ((j & 4) << 2) + hlf * 8 + ((j & 3) << 1);
    f.u[j] = *(const unsigned int*)(row + k);
  }
  return f.v;
}

// B fragment (32x16 bf16, K x N) from an N-major buffer: src[n*ld + k].
// lane&15 = N, K layout same as A: VGPR j<4 -> K = half*8 + 2j ; j>=4 -> K = 16 + half*8 + 2(j-4).
__device__ __forceinline__ v16bf frag_b_from(const unsigned short* src, int ld, int lane) {
  const int n = lane & 15, hlf = lane >> 4;
  const unsigned short* row = src + n * ld;
  FragBF f;
#pragma unroll
  for (int j = 0; j < 8; ++j) f.u[j] = *(const unsigned int*)(row + ((j & 4) << 2) + hlf * 8 + ((j & 3) << 1));
  return f.v;
}

__device__ __forceinline__ v8f wmma_bf16(v16bf a, v16bf b, v8f c) {
  return __builtin_amdgcn_wmma_f32_16x16x32_bf16(false, a, false, b, (short)0, c, false, false);
}

// ---------------- weight f32 -> bf16 ----------------
__global__ __launch_bounds__(256) void cvt_f32_bf16(const float* __restrict__ in,
                                                    unsigned short* __restrict__ out, int n) {
  int i = blockIdx.x * 256 + threadIdx.x;
  if (i < n) out[i] = f2bf(in[i]);
}

// ---------------- QKV projection ----------------
// X[4096,1024]f32 @ W[1024,1024]bf16 + b -> bf16.
// transp==0: head-split [B,H,S,HD] (Q, K).  transp==1: [B,H,HD,S] (V, n-major for PV B-operand).
__global__ __launch_bounds__(256) void gemm_qkv(const float* __restrict__ X,
                                                const unsigned short* __restrict__ Wb,
                                                const float* __restrict__ bias,
                                                unsigned short* __restrict__ Out,
                                                int transp) {
  __shared__ unsigned short As[BM * LDA];
  __shared__ unsigned short Bs[BN * LDB];
  const int tid = threadIdx.x, lane = tid & 31, w = tid >> 5;
  const int m0 = blockIdx.x * BM, n0 = blockIdx.y * BN;

  v8f acc[4] = {};
  for (int k0 = 0; k0 < D_DIM; k0 += BK) {
    // A tile 128x32 f32 -> bf16 LDS
#pragma unroll
    for (int i = 0; i < 4; ++i) {
      int idx = tid + i * 256;            // 1024 float4 slots
      int row = idx >> 3, qc = idx & 7;
      float4 f4 = *(const float4*)(X + (size_t)(m0 + row) * D_DIM + k0 + qc * 4);
      unsigned short* dst = As + row * LDA + qc * 4;
      dst[0] = f2bf(f4.x); dst[1] = f2bf(f4.y); dst[2] = f2bf(f4.z); dst[3] = f2bf(f4.w);
    }
    // B tile 32x64 bf16 -> N-major LDS
#pragma unroll
    for (int i = 0; i < 4; ++i) {
      int idx = tid + i * 256;            // 1024 dword slots
      int kr = idx >> 5, cp = idx & 31;
      unsigned int u = *(const unsigned int*)(Wb + (size_t)(k0 + kr) * D_DIM + n0 + cp * 2);
      Bs[(cp * 2 + 0) * LDB + kr] = (unsigned short)(u & 0xFFFFu);
      Bs[(cp * 2 + 1) * LDB + kr] = (unsigned short)(u >> 16);
    }
    __syncthreads();
    v16bf a = frag_a_from(As + (w * 16) * LDA, LDA, lane);
#pragma unroll
    for (int nt = 0; nt < 4; ++nt) {
      v16bf b = frag_b_from(Bs + (nt * 16) * LDB, LDB, lane);
      acc[nt] = wmma_bf16(a, b, acc[nt]);
    }
    __syncthreads();
  }
  // epilogue: +bias, bf16
  const int hlf = lane >> 4, nn = lane & 15;
#pragma unroll
  for (int nt = 0; nt < 4; ++nt) {
    int n = n0 + nt * 16 + nn;
    int h = n >> 6, hd = n & 63;
    float bv = bias[n];
#pragma unroll
    for (int r = 0; r < 8; ++r) {
      int m = m0 + w * 16 + hlf * 8 + r;
      int bb = m >> 11, s = m & 2047;
      unsigned short val = f2bf(acc[nt][r] + bv);
      if (transp)
        Out[((size_t)(bb * NHEAD + h) * HDIM + hd) * S_LEN + s] = val;   // [B,H,HD,S]
      else
        Out[((size_t)(bb * NHEAD + h) * S_LEN + s) * HDIM + hd] = val;   // [B,H,S,HD]
    }
  }
}

// ---------------- fully fused scores + softmax + P@V ----------------
// One block = 16 query rows x ALL 2048 keys for one (b,h).
// 1) Wave w computes scores for keys [w*256, w*256+256) -> 16 acc tiles (128 VGPRs).
// 2) Row softmax: in-lane partials, __shfl_xor across the 16-lane half, LDS across waves.
// 3) Normalized P: written fp32 to d_out (mandatory output, exactly once) AND staged
//    bf16 into LDS in two 1024-key rounds; each round all 8 waves accumulate WMMA
//    partials of P@V over a 128-key slice against Vt ([B,H,HD,S], n-major, L2-resident).
// 4) Cross-wave reduction of 8 partial 16x64 tiles in (aliased) LDS -> AO bf16.
__global__ __launch_bounds__(256) void attn_fused(
    const unsigned short* __restrict__ Qb,
    const unsigned short* __restrict__ Kb,
    const unsigned short* __restrict__ Vt,
    float* __restrict__ P,
    unsigned short* __restrict__ AO) {
  __shared__ unsigned short Pl[16 * LDP];   // 33 KB: staged P rows / aliased reduce buf
  __shared__ float red[8][16];
  const int bh = blockIdx.y, bb = bh >> 4, h = bh & 15;
  const int m0 = blockIdx.x * 16;
  const int tid = threadIdx.x, lane = tid & 31, w = tid >> 5;
  const int hlf = lane >> 4, nn = lane & 15;
  const unsigned short* Q = Qb + (size_t)bh * S_LEN * HDIM;
  const unsigned short* K = Kb + (size_t)bh * S_LEN * HDIM;
  const unsigned short* V = Vt + (size_t)bh * HDIM * S_LEN;
  float* Pout = P + (size_t)bh * S_LEN * S_LEN + (size_t)m0 * S_LEN;

  // ---- scores: 16 x 256 per wave ----
  v8f acc[16] = {};
  const v16bf a0 = frag_a_from(Q + (size_t)m0 * HDIM, HDIM, lane);       // K=0..31
  const v16bf a1 = frag_a_from(Q + (size_t)m0 * HDIM + 32, HDIM, lane);  // K=32..63
#pragma unroll
  for (int t = 0; t < 16; ++t) {
    const unsigned short* Krow = K + (size_t)(w * 256 + t * 16) * HDIM;
    v16bf b0 = frag_b_from(Krow, HDIM, lane);
    v16bf b1 = frag_b_from(Krow + 32, HDIM, lane);
    acc[t] = wmma_bf16(a0, b0, acc[t]);
    acc[t] = wmma_bf16(a1, b1, acc[t]);
  }
#pragma unroll
  for (int t = 0; t < 16; ++t)
#pragma unroll
    for (int r = 0; r < 8; ++r) acc[t][r] *= 0.125f;     // 1/sqrt(64), before max

  // ---- row max ----
  float pm[8];
#pragma unroll
  for (int r = 0; r < 8; ++r) {
    float m = acc[0][r];
#pragma unroll
    for (int t = 1; t < 16; ++t) m = fmaxf(m, acc[t][r]);
    pm[r] = m;
  }
#pragma unroll
  for (int d = 1; d < 16; d <<= 1)
#pragma unroll
    for (int r = 0; r < 8; ++r) pm[r] = fmaxf(pm[r], __shfl_xor(pm[r], d, 32));
  if (nn == 0) {
#pragma unroll
    for (int r = 0; r < 8; ++r) red[w][hlf * 8 + r] = pm[r];
  }
  __syncthreads();
  float rmax[8];
#pragma unroll
  for (int r = 0; r < 8; ++r) {
    float m = red[0][hlf * 8 + r];
#pragma unroll
    for (int ww = 1; ww < 8; ++ww) m = fmaxf(m, red[ww][hlf * 8 + r]);
    rmax[r] = m;
  }
  __syncthreads();

  // ---- exp + row sum ----
  float ps[8];
#pragma unroll
  for (int r = 0; r < 8; ++r) ps[r] = 0.0f;
#pragma unroll
  for (int t = 0; t < 16; ++t)
#pragma unroll
    for (int r = 0; r < 8; ++r) {
      float e = __expf(acc[t][r] - rmax[r]);
      acc[t][r] = e;
      ps[r] += e;
    }
#pragma unroll
  for (int d = 1; d < 16; d <<= 1)
#pragma unroll
    for (int r = 0; r < 8; ++r) ps[r] += __shfl_xor(ps[r], d, 32);
  if (nn == 0) {
#pragma unroll
    for (int r = 0; r < 8; ++r) red[w][hlf * 8 + r] = ps[r];
  }
  __syncthreads();
  float rinv[8];
#pragma unroll
  for (int r = 0; r < 8; ++r) {
    float s = red[0][hlf * 8 + r];
#pragma unroll
    for (int ww = 1; ww < 8; ++ww) s += red[ww][hlf * 8 + r];
    rinv[r] = 1.0f / s;
  }

  // ---- normalize in regs, write fp32 P once, and run P@V in two staged rounds ----
#pragma unroll
  for (int t = 0; t < 16; ++t)
#pragma unroll
    for (int r = 0; r < 8; ++r) acc[t][r] *= rinv[r];

  // mandatory attn_weights output (only pass over P)
#pragma unroll
  for (int t = 0; t < 16; ++t) {
    int n = w * 256 + t * 16 + nn;
#pragma unroll
    for (int r = 0; r < 8; ++r)
      Pout[(size_t)(hlf * 8 + r) * S_LEN + n] = acc[t][r];
  }

  v8f oacc[4] = {};
#pragma unroll
  for (int round = 0; round < 2; ++round) {
    __syncthreads();                        // Pl free / previous round consumed
    if ((w >> 2) == round) {                // waves owning keys [round*1024, +1024)
      int base = (w & 3) * 256;             // local key offset of this wave's chunk
#pragma unroll
      for (int t = 0; t < 16; ++t) {
        int ln = base + t * 16 + nn;
#pragma unroll
        for (int r = 0; r < 8; ++r)
          Pl[(hlf * 8 + r) * LDP + ln] = f2bf(acc[t][r]);
      }
    }
    __syncthreads();
    // all 8 waves: 128-key slice of this round's 1024 staged keys
#pragma unroll
    for (int kk = 0; kk < 4; ++kk) {
      int lk = w * 128 + kk * 32;                          // local key
      v16bf pa = frag_a_from(Pl + lk, LDP, lane);
      int gk = round * 1024 + lk;                          // global key
#pragma unroll
      for (int nt = 0; nt < 4; ++nt) {
        v16bf vb = frag_b_from(V + (size_t)(nt * 16) * S_LEN + gk, S_LEN, lane);
        oacc[nt] = wmma_bf16(pa, vb, oacc[nt]);
      }
    }
  }

  // ---- cross-wave reduce 8 partial 16x64 tiles (alias Pl as float buf) ----
  __syncthreads();                          // done reading Pl as staged P
  float* redav = (float*)Pl;                // [8][16][64] = 32 KB <= sizeof(Pl)
#pragma unroll
  for (int nt = 0; nt < 4; ++nt) {
    int hd = nt * 16 + nn;
#pragma unroll
    for (int r = 0; r < 8; ++r)
      redav[((size_t)w * 16 + hlf * 8 + r) * 64 + hd] = oacc[nt][r];
  }
  __syncthreads();
#pragma unroll
  for (int e = 0; e < 4; ++e) {
    int idx = tid + e * 256;                // 1024 outputs: 16 rows x 64 cols
    int m = idx >> 6, hd = idx & 63;
    float s = 0.0f;
#pragma unroll
    for (int ww = 0; ww < 8; ++ww) s += redav[((size_t)ww * 16 + m) * 64 + hd];
    AO[(size_t)(bb * S_LEN + m0 + m) * D_DIM + h * HDIM + hd] = f2bf(s);
  }
}

// ---------------- output projection: AO[4096,1024]bf16 @ Wo bf16 + bo -> f32 ----------------
__global__ __launch_bounds__(256) void gemm_out(const unsigned short* __restrict__ A,
                                                const unsigned short* __restrict__ Wb,
                                                const float* __restrict__ bias,
                                                float* __restrict__ Out) {
  __shared__ unsigned short As[BM * LDA];
  __shared__ unsigned short Bs[BN * LDB];
  const int tid = threadIdx.x, lane = tid & 31, w = tid >> 5;
  const int m0 = blockIdx.x * BM, n0 = blockIdx.y * BN;

  v8f acc[4] = {};
  for (int k0 = 0; k0 < D_DIM; k0 += BK) {
#pragma unroll
    for (int i = 0; i < 4; ++i) {
      int idx = tid + i * 256;
      int row = idx >> 3, qc = idx & 7;
      const unsigned int* src = (const unsigned int*)(A + (size_t)(m0 + row) * D_DIM + k0);
      *(unsigned int*)(As + row * LDA + qc * 4)     = src[qc * 2];
      *(unsigned int*)(As + row * LDA + qc * 4 + 2) = src[qc * 2 + 1];
    }
#pragma unroll
    for (int i = 0; i < 4; ++i) {
      int idx = tid + i * 256;
      int kr = idx >> 5, cp = idx & 31;
      unsigned int u = *(const unsigned int*)(Wb + (size_t)(k0 + kr) * D_DIM + n0 + cp * 2);
      Bs[(cp * 2 + 0) * LDB + kr] = (unsigned short)(u & 0xFFFFu);
      Bs[(cp * 2 + 1) * LDB + kr] = (unsigned short)(u >> 16);
    }
    __syncthreads();
    v16bf a = frag_a_from(As + (w * 16) * LDA, LDA, lane);
#pragma unroll
    for (int nt = 0; nt < 4; ++nt) {
      v16bf b = frag_b_from(Bs + (nt * 16) * LDB, LDB, lane);
      acc[nt] = wmma_bf16(a, b, acc[nt]);
    }
    __syncthreads();
  }
  const int hlf = lane >> 4, nn = lane & 15;
#pragma unroll
  for (int nt = 0; nt < 4; ++nt) {
    int n = n0 + nt * 16 + nn;
    float bv = bias[n];
#pragma unroll
    for (int r = 0; r < 8; ++r) {
      int m = m0 + w * 16 + hlf * 8 + r;
      Out[(size_t)m * D_DIM + n] = acc[nt][r] + bv;
    }
  }
}

extern "C" void kernel_launch(void* const* d_in, const int* in_sizes, int n_in,
                              void* d_out, int out_size, void* d_ws, size_t ws_size,
                              hipStream_t stream) {
  const float* query = (const float*)d_in[0];
  const float* key_  = (const float*)d_in[1];
  const float* value = (const float*)d_in[2];
  const float* Wq = (const float*)d_in[3];
  const float* bq = (const float*)d_in[4];
  const float* Wk = (const float*)d_in[5];
  const float* bk = (const float*)d_in[6];
  const float* Wv = (const float*)d_in[7];
  const float* bv = (const float*)d_in[8];
  const float* Wo = (const float*)d_in[9];
  const float* bo = (const float*)d_in[10];

  float* out  = (float*)d_out;                          // [B,S,D]
  float* attn = out + (size_t)2 * S_LEN * D_DIM;        // [B,H,S,S]

  unsigned short* Wqb = (unsigned short*)d_ws;
  unsigned short* Wkb = Wqb + (size_t)D_DIM * D_DIM;
  unsigned short* Wvb = Wkb + (size_t)D_DIM * D_DIM;
  unsigned short* Wob = Wvb + (size_t)D_DIM * D_DIM;
  unsigned short* Qb  = Wob + (size_t)D_DIM * D_DIM;    // [B,H,S,64]
  unsigned short* Kb  = Qb + (size_t)MROWS * D_DIM;     // [B,H,S,64]
  unsigned short* Vtb = Kb + (size_t)MROWS * D_DIM;     // [B,H,64,S] transposed
  unsigned short* AOb = Vtb + (size_t)MROWS * D_DIM;    // [B,S,D] bf16

  const int nW = D_DIM * D_DIM;
  cvt_f32_bf16<<<nW / 256, 256, 0, stream>>>(Wq, Wqb, nW);
  cvt_f32_bf16<<<nW / 256, 256, 0, stream>>>(Wk, Wkb, nW);
  cvt_f32_bf16<<<nW / 256, 256, 0, stream>>>(Wv, Wvb, nW);
  cvt_f32_bf16<<<nW / 256, 256, 0, stream>>>(Wo, Wob, nW);

  dim3 gproj(MROWS / BM, D_DIM / BN);                   // 32 x 16
  gemm_qkv<<<gproj, 256, 0, stream>>>(query, Wqb, bq, Qb, 0);
  gemm_qkv<<<gproj, 256, 0, stream>>>(key_,  Wkb, bk, Kb, 0);
  gemm_qkv<<<gproj, 256, 0, stream>>>(value, Wvb, bv, Vtb, 1);

  // fused scores + softmax + P@V: P written once, never re-read from HBM
  attn_fused<<<dim3(S_LEN / 16, BH), 256, 0, stream>>>(Qb, Kb, Vtb, attn, AOb);

  gemm_out<<<gproj, 256, 0, stream>>>(AOb, Wob, bo, out);
}
